// Attention_local_49907519980096
// MI455X (gfx1250) — matrix operand order repaired and checked by reading, not compiled, not run on hardware
//
#include <hip/hip_runtime.h>
#include <math.h>

// ---------------------------------------------------------------------------
// Problem constants (match reference)
// ---------------------------------------------------------------------------
#define BB     2
#define DIM    64
#define HH     256
#define WW     256
#define HPD    128      // prob/entropy H
#define WPD    128      // prob/entropy W
#define CCH    2
#define WIN    16
#define WIN2   8
#define KEEP1  120
#define KEEP2  30
#define NW     121              // window grid per axis
#define NBOX   (NW * NW)        // 14641 candidate boxes
#define NIMG   (BB * KEEP2)     // 60 window "images"
#define BN_SCALE 0.99999500003749973f   // 1/sqrt(1+1e-5)

typedef __attribute__((ext_vector_type(16))) _Float16 v16h;
typedef __attribute__((ext_vector_type(8)))  _Float16 v8h;
typedef __attribute__((ext_vector_type(8)))  float    v8f;

union F16x16 { v16h v; struct { v8h lo, hi; } p; };

// ---------------------------------------------------------------------------
// Weight pre-pack: fp32 OIHW -> f16 B-fragment layout
//   wp[ntile][chunk][tap][lane][e]  (lane = wave lane, e = fragment element)
//   B fragment K-mapping: K = (lane>>4)*8 + (e&7) + (e>>3)*16 ; column N = lane&15
// ---------------------------------------------------------------------------
__global__ void pack_weights_kernel(const float* __restrict__ w, _Float16* __restrict__ wp,
                                    int Cout, int Cin)
{
    int i = blockIdx.x * blockDim.x + threadIdx.x;
    int total = Cout * Cin * 9;
    if (i >= total) return;
    int e = i & 15;  int tmp = i >> 4;
    int lane = tmp & 31;  tmp >>= 5;
    int t = tmp % 9;  tmp /= 9;
    int nchunks = Cin >> 5;
    int chunk = tmp % nchunks;
    int ntile = tmp / nchunks;
    int cout = ntile * 16 + (lane & 15);
    int cin  = chunk * 32 + (lane >> 4) * 8 + (e & 7) + (e >> 3) * 16;
    wp[i] = (_Float16)w[((size_t)cout * Cin + cin) * 9 + t];
}

// ---------------------------------------------------------------------------
// 3x3 conv (pad 1) + bias + BN + ReLU, f16 WMMA implicit GEMM, Cout = 64.
// Activations: f16 channel-last [img][y][x][CIN]; weights pre-packed.
// One wave computes 16 pixels x 64 out channels (4 accumulator fragments):
// per (chunk, tap): 1 A build (2x b128 + pad-select) shared by 4 WMMAs
// (8x b128 B loads) -> 2.5 loads per WMMA, 4 independent back-to-back WMMAs.
// ---------------------------------------------------------------------------
template <int CIN, int H, int W, bool F32OUT>
__global__ __launch_bounds__(32)
void conv3x3_wmma_kernel(const _Float16* __restrict__ in, const _Float16* __restrict__ wpk,
                         const float* __restrict__ bias, void* __restrict__ outp,
                         int outCtot, int outCoff)
{
    const int lane = threadIdx.x;            // wave32, all lanes active
    const int mcol = lane & 15;              // A row / B,D column
    const int kh   = lane >> 4;              // K-half selector
    constexpr int tilesPerRow = W >> 4;
    constexpr int NCH = CIN >> 5;            // 32-channel chunks
    const int y   = blockIdx.x / tilesPerRow;
    const int x0  = (blockIdx.x % tilesPerRow) << 4;
    const int xA  = x0 + mcol;
    const int img = blockIdx.z;
    const _Float16* inImg = in + (size_t)img * H * W * CIN;
    const v8h zeroh = (v8h)(_Float16)0.0f;

    v8f acc[4] = {{}, {}, {}, {}};
    for (int c0 = 0; c0 < CIN; c0 += 32) {
#pragma unroll
        for (int t = 0; t < 9; ++t) {
            const int r  = t / 3;
            const int s  = t - 3 * r;
            const int yy = y + r - 1;
            const int xx = xA + s - 1;
            const bool inb = (yy >= 0) && (yy < H) && (xx >= 0) && (xx < W);
            const int yyc = min(max(yy, 0), H - 1);
            const int xxc = min(max(xx, 0), W - 1);
            const _Float16* ap = inImg + ((size_t)yyc * W + xxc) * CIN + c0 + kh * 8;
            v8h alo = *(const v8h*)(ap);       // channels c0+8*kh    .. +7
            v8h ahi = *(const v8h*)(ap + 16);  // channels c0+16+8*kh .. +7
            F16x16 a;
            a.p.lo = inb ? alo : zeroh;
            a.p.hi = inb ? ahi : zeroh;
#pragma unroll
            for (int k = 0; k < 4; ++k) {      // 4 out-channel tiles share A
                const _Float16* bp =
                    wpk + ((((size_t)k * NCH + (c0 >> 5)) * 9 + t) * 32 + lane) * 16;
                F16x16 b;
                b.p.lo = *(const v8h*)(bp);
                b.p.hi = *(const v8h*)(bp + 8);
                acc[k] = __builtin_amdgcn_wmma_f32_16x16x32_f16(
                    false, a.v, false, b.v, (short)0, acc[k], false, false);
            }
        }
    }
    // D layout: lane holds column N=mcol (out channel), rows v+8*kh (x pixels)
    if (F32OUT) {
        float* out = (float*)outp;
#pragma unroll
        for (int k = 0; k < 4; ++k) {
            const int ch = outCoff + k * 16 + mcol;
            const float bia = bias[k * 16 + mcol];
            float* op = out + (((size_t)img * outCtot + ch) * H + y) * (size_t)W
                        + x0 + 8 * kh;
#pragma unroll
            for (int v = 0; v < 8; ++v) {
                float o = (acc[k][v] + bia) * BN_SCALE;
                op[v] = fmaxf(o, 0.0f);
            }
        }
    } else {
        _Float16* out = (_Float16*)outp;
        size_t pixbase = (((size_t)img * H + y) * W + x0 + 8 * kh) * outCtot + outCoff;
#pragma unroll
        for (int k = 0; k < 4; ++k) {
            const float bia = bias[k * 16 + mcol];
            size_t base = pixbase + k * 16 + mcol;
#pragma unroll
            for (int v = 0; v < 8; ++v) {
                float o = (acc[k][v] + bia) * BN_SCALE;
                out[base + (size_t)v * outCtot] = (_Float16)fmaxf(o, 0.0f);
            }
        }
    }
}

// ---------------------------------------------------------------------------
// Entropy of prob and y over the channel axis (C=2)
// ---------------------------------------------------------------------------
__global__ void entropy_kernel(const float* __restrict__ prob, const float* __restrict__ yv,
                               float* __restrict__ ent, float* __restrict__ enty)
{
    int i = blockIdx.x * blockDim.x + threadIdx.x;
    const int n = BB * HPD * WPD;
    if (i >= n) return;
    int b = i / (HPD * WPD);
    int p = i - b * HPD * WPD;
    float e = 0.f, ey = 0.f;
    for (int c = 0; c < CCH; ++c) {
        float v = prob[((size_t)b * CCH + c) * HPD * WPD + p];
        e -= v * log2f(v + 1e-10f);
        float u = yv[((size_t)b * CCH + c) * HPD * WPD + p];
        ey -= u * log2f(u + 1e-10f);
    }
    ent[i] = e;
    enty[i] = ey;
}

// 8x8 box filter with fix_w over entropy -> (121,121) window scores
__global__ void window_score_kernel(const float* __restrict__ ent,
                                    const float* __restrict__ fixw,
                                    float* __restrict__ score)
{
    int i = blockIdx.x * blockDim.x + threadIdx.x;
    if (i >= BB * NBOX) return;
    int b  = i / NBOX;
    int rc = i - b * NBOX;
    int r  = rc / NW;
    int c  = rc - r * NW;
    const float* e = ent + (size_t)b * HPD * WPD;
    float s = 0.f;
    for (int u = 0; u < WIN2; ++u)
        for (int v = 0; v < WIN2; ++v)
            s += e[(r + u) * WPD + (c + v)] * fixw[u * WIN2 + v];
    score[i] = s * (1.0f / (WIN2 * WIN2));
}

// ---------------------------------------------------------------------------
// NMS pass 1: 120 sequential argmax+suppress rounds over 14641 boxes.
// ---------------------------------------------------------------------------
__global__ __launch_bounds__(256)
void nms1_kernel(const float* __restrict__ score, int* __restrict__ keep1)
{
    const int b   = blockIdx.x;
    const int tid = threadIdx.x;
    __shared__ unsigned char supp[NBOX];
    __shared__ float sv[256];
    __shared__ int   si[256];
    __shared__ int   selj;
    for (int i = tid; i < NBOX; i += 256) supp[i] = 0;
    __syncthreads();
    const float* sc = score + (size_t)b * NBOX;
    for (int it = 0; it < KEEP1; ++it) {
        float best = -INFINITY;
        int   bi   = 0x7fffffff;
        for (int i = tid; i < NBOX; i += 256) {
            float v = supp[i] ? -INFINITY : sc[i];
            if (v > best || (v == best && i < bi)) { best = v; bi = i; }
        }
        sv[tid] = best; si[tid] = bi;
        __syncthreads();
        for (int s = 128; s > 0; s >>= 1) {
            if (tid < s) {
                if (sv[tid + s] > sv[tid] ||
                    (sv[tid + s] == sv[tid] && si[tid + s] < si[tid])) {
                    sv[tid] = sv[tid + s]; si[tid] = si[tid + s];
                }
            }
            __syncthreads();
        }
        if (tid == 0) { selj = si[0]; keep1[b * KEEP1 + it] = si[0]; }
        __syncthreads();
        const int j = selj;
        const float x1j = 2.0f * (float)(j % NW);
        const float y1j = 2.0f * (float)(j / NW);
        for (int i = tid; i < NBOX; i += 256) {
            float x1 = 2.0f * (float)(i % NW);
            float y1 = 2.0f * (float)(i / NW);
            float iw = fmaxf(fminf(x1 + 15.f, x1j + 15.f) - fmaxf(x1, x1j), 0.f);
            float ih = fmaxf(fminf(y1 + 15.f, y1j + 15.f) - fmaxf(y1, y1j), 0.f);
            float inter = iw * ih;                 // all areas = 15*15 = 225
            float iou = inter / (450.f - inter);
            if (iou > 0.4f) supp[i] = 1;
        }
        __syncthreads();
    }
}

// Re-score the 120 kept boxes on entropy_y (reference uses (x/2, y/2) as
// (row, col) in dynamic_slice -> transposed access kept faithful).
__global__ void crop_score_kernel(const float* __restrict__ enty,
                                  const float* __restrict__ fixw,
                                  const int* __restrict__ keep1,
                                  float* __restrict__ wsc)
{
    int i = blockIdx.x * blockDim.x + threadIdx.x;
    if (i >= BB * KEEP1) return;
    int b = i / KEEP1;
    int j = keep1[i];
    int c = j % NW;   // x1/2 -> row start
    int r = j / NW;   // y1/2 -> col start
    const float* e = enty + (size_t)b * HPD * WPD;
    float s = 0.f;
    for (int u = 0; u < WIN2; ++u)
        for (int v = 0; v < WIN2; ++v)
            s += e[(c + u) * WPD + (r + v)] * fixw[u * WIN2 + v];
    wsc[i] = s * (1.0f / (WIN2 * WIN2));
}

// NMS pass 2: 30 rounds over the 120 kept boxes, thr 0.2; emits (sx, sy).
__global__ __launch_bounds__(128)
void nms2_kernel(const float* __restrict__ wsc, const int* __restrict__ keep1,
                 int* __restrict__ sxA, int* __restrict__ syA)
{
    const int b   = blockIdx.x;
    const int tid = threadIdx.x;   // 128 threads
    __shared__ unsigned char supp[KEEP1];
    __shared__ float sv[128];
    __shared__ int   si[128];
    __shared__ int   selt;
    if (tid < KEEP1) supp[tid] = 0;
    __syncthreads();
    for (int it = 0; it < KEEP2; ++it) {
        float best = -INFINITY;
        int   bi   = 0x7fffffff;
        if (tid < KEEP1) {
            best = supp[tid] ? -INFINITY : wsc[b * KEEP1 + tid];
            bi   = tid;
        }
        sv[tid] = best; si[tid] = bi;
        __syncthreads();
        for (int s = 64; s > 0; s >>= 1) {
            if (tid < s) {
                if (sv[tid + s] > sv[tid] ||
                    (sv[tid + s] == sv[tid] && si[tid + s] < si[tid])) {
                    sv[tid] = sv[tid + s]; si[tid] = si[tid + s];
                }
            }
            __syncthreads();
        }
        if (tid == 0) {
            int t = si[0];
            selt = t;
            int j = keep1[b * KEEP1 + t];
            sxA[b * KEEP2 + it] = 2 * (j % NW);
            syA[b * KEEP2 + it] = 2 * (j / NW);
        }
        __syncthreads();
        const int jj  = keep1[b * KEEP1 + selt];
        const float x1j = 2.0f * (float)(jj % NW);
        const float y1j = 2.0f * (float)(jj / NW);
        if (tid < KEEP1) {
            int ji = keep1[b * KEEP1 + tid];
            float x1 = 2.0f * (float)(ji % NW);
            float y1 = 2.0f * (float)(ji / NW);
            float iw = fmaxf(fminf(x1 + 15.f, x1j + 15.f) - fmaxf(x1, x1j), 0.f);
            float ih = fmaxf(fminf(y1 + 15.f, y1j + 15.f) - fmaxf(y1, y1j), 0.f);
            float inter = iw * ih;
            float iou = inter / (450.f - inter);
            if (iou > 0.2f) supp[tid] = 1;
        }
        __syncthreads();
    }
}

// Gather (fp32 NCHW copy for the MLE VALU path)
__global__ void gather_windows32_kernel(const float* __restrict__ x,
                                        const int* __restrict__ sxA,
                                        const int* __restrict__ syA,
                                        float* __restrict__ wf32)
{
    int i = blockIdx.x * blockDim.x + threadIdx.x;
    const int total = NIMG * DIM * WIN * WIN;
    if (i >= total) return;
    int v = i % WIN; int tmp = i / WIN;
    int u = tmp % WIN; tmp /= WIN;
    int ch = tmp % DIM; tmp /= DIM;
    int t = tmp % KEEP2;
    int b = tmp / KEEP2;
    int sy = syA[b * KEEP2 + t], sx = sxA[b * KEEP2 + t];
    wf32[i] = x[(((size_t)b * DIM + ch) * HH + (sy + u)) * WW + (sx + v)];
}

// Gather (f16 channel-last copy for the WMMA path): wf16[img][u][v][ch]
__global__ void gather_windows16_kernel(const float* __restrict__ x,
                                        const int* __restrict__ sxA,
                                        const int* __restrict__ syA,
                                        _Float16* __restrict__ wf16)
{
    int i = blockIdx.x * blockDim.x + threadIdx.x;
    const int total = NIMG * WIN * WIN * DIM;
    if (i >= total) return;
    int ch = i % DIM; int tmp = i / DIM;
    int v = tmp % WIN; tmp /= WIN;
    int u = tmp % WIN; tmp /= WIN;
    int t = tmp % KEEP2;
    int b = tmp / KEEP2;
    int sy = syA[b * KEEP2 + t], sx = sxA[b * KEEP2 + t];
    wf16[i] = (_Float16)x[(((size_t)b * DIM + ch) * HH + (sy + u)) * WW + (sx + v)];
}

// ---------------------------------------------------------------------------
// MLE block (small tensors, VALU kernels). Planes are 16x16 = 256 pixels.
// ---------------------------------------------------------------------------
__global__ void mle_c1_kernel(const float* __restrict__ wf, const float* __restrict__ w,
                              const float* __restrict__ bia, float* __restrict__ xc)
{
    int i = blockIdx.x * blockDim.x + threadIdx.x;
    const int total = NIMG * 16 * 256;
    if (i >= total) return;
    int p = i % 256; int tmp = i / 256;
    int co = tmp % 16; int img = tmp / 16;
    float s = bia[co];
    for (int ci = 0; ci < DIM; ++ci)
        s += wf[((size_t)img * DIM + ci) * 256 + p] * w[co * DIM + ci];
    xc[i] = fmaxf(s * BN_SCALE, 0.0f);
}

// d1: 1x7 conv (width, pad 3), 16->8, relu(bn(.)) into cat4 channels [0,8)
__global__ void mle_d1_kernel(const float* __restrict__ xc, const float* __restrict__ w,
                              const float* __restrict__ bia, float* __restrict__ cat4)
{
    int i = blockIdx.x * blockDim.x + threadIdx.x;
    const int total = NIMG * 8 * 256;
    if (i >= total) return;
    int p = i % 256; int tmp = i / 256;
    int co = tmp % 8; int img = tmp / 8;
    int y = p / 16, x = p % 16;
    float s = bia[co];
    for (int ci = 0; ci < 16; ++ci)
        for (int ks = 0; ks < 7; ++ks) {
            int xx = x + ks - 3;
            if (xx >= 0 && xx < 16)
                s += xc[((size_t)img * 16 + ci) * 256 + y * 16 + xx] * w[(co * 16 + ci) * 7 + ks];
        }
    cat4[((size_t)img * 32 + 0 + co) * 256 + p] = fmaxf(s * BN_SCALE, 0.0f);
}

// d2: 7x1 conv (height, pad 3), 16->8, into cat4 channels [8,16)
__global__ void mle_d2_kernel(const float* __restrict__ xc, const float* __restrict__ w,
                              const float* __restrict__ bia, float* __restrict__ cat4)
{
    int i = blockIdx.x * blockDim.x + threadIdx.x;
    const int total = NIMG * 8 * 256;
    if (i >= total) return;
    int p = i % 256; int tmp = i / 256;
    int co = tmp % 8; int img = tmp / 8;
    int y = p / 16, x = p % 16;
    float s = bia[co];
    for (int ci = 0; ci < 16; ++ci)
        for (int kr = 0; kr < 7; ++kr) {
            int yy = y + kr - 3;
            if (yy >= 0 && yy < 16)
                s += xc[((size_t)img * 16 + ci) * 256 + yy * 16 + x] * w[(co * 16 + ci) * 7 + kr];
        }
    cat4[((size_t)img * 32 + 8 + co) * 256 + p] = fmaxf(s * BN_SCALE, 0.0f);
}

// h-shear: ht plane (16,31); flat index q: j=q/32, pos=q%32; value if pos<16
__global__ void htr_kernel(const float* __restrict__ xc, float* __restrict__ ht)
{
    int i = blockIdx.x * blockDim.x + threadIdx.x;
    const int total = NIMG * 16 * 496;
    if (i >= total) return;
    int q = i % 496; int pc = i / 496;
    int j = q >> 5, pos = q & 31;
    ht[i] = (pos < 16) ? xc[(size_t)pc * 256 + j * 16 + pos] : 0.0f;
}

// d3: 7x1 (height, pad 3) on the (16,31) sheared planes, 16->8 (+bias)
__global__ void mle_d3_kernel(const float* __restrict__ ht, const float* __restrict__ w,
                              const float* __restrict__ bia, float* __restrict__ d3o)
{
    int i = blockIdx.x * blockDim.x + threadIdx.x;
    const int total = NIMG * 8 * 496;
    if (i >= total) return;
    int q = i % 496; int tmp = i / 496;
    int co = tmp % 8; int img = tmp / 8;
    int y = q / 31, k = q % 31;
    float s = bia[co];
    for (int ci = 0; ci < 16; ++ci)
        for (int kr = 0; kr < 7; ++kr) {
            int yy = y + kr - 3;
            if (yy >= 0 && yy < 16)
                s += ht[((size_t)img * 16 + ci) * 496 + yy * 31 + k] * w[(co * 16 + ci) * 7 + kr];
        }
    d3o[i] = s;
}

// inverse h-shear: out(y,k) = d3o_flat[y*32+k]; relu(bn) into cat4 [16,24)
__global__ void mle_invh_kernel(const float* __restrict__ d3o, float* __restrict__ cat4)
{
    int i = blockIdx.x * blockDim.x + threadIdx.x;
    const int total = NIMG * 8 * 256;
    if (i >= total) return;
    int p = i % 256; int tmp = i / 256;
    int co = tmp % 8; int img = tmp / 8;
    int y = p / 16, k = p % 16;
    int f = y * 32 + k;   // < 496 always
    float v = d3o[((size_t)img * 8 + co) * 496 + f];
    cat4[((size_t)img * 32 + 16 + co) * 256 + p] = fmaxf(v * BN_SCALE, 0.0f);
}

// v-shear: vt plane (31,16): vt[a][b] = x[pos][j] with f=b*31+a, j=f/32, pos=f%32
__global__ void vtr_kernel(const float* __restrict__ xc, float* __restrict__ vt)
{
    int i = blockIdx.x * blockDim.x + threadIdx.x;
    const int total = NIMG * 16 * 496;
    if (i >= total) return;
    int q = i % 496; int pc = i / 496;
    int a = q / 16, bcol = q % 16;
    int f = bcol * 31 + a;
    int j = f >> 5, pos = f & 31;
    vt[i] = (pos < 16) ? xc[(size_t)pc * 256 + pos * 16 + j] : 0.0f;
}

// d4: 1x7 (width, pad 3) on the (31,16) sheared planes, 16->8 (+bias)
__global__ void mle_d4_kernel(const float* __restrict__ vt, const float* __restrict__ w,
                              const float* __restrict__ bia, float* __restrict__ d4o)
{
    int i = blockIdx.x * blockDim.x + threadIdx.x;
    const int total = NIMG * 8 * 496;
    if (i >= total) return;
    int q = i % 496; int tmp = i / 496;
    int co = tmp % 8; int img = tmp / 8;
    int a = q / 16, bcol = q % 16;
    float s = bia[co];
    for (int ci = 0; ci < 16; ++ci)
        for (int ks = 0; ks < 7; ++ks) {
            int bs = bcol + ks - 3;
            if (bs >= 0 && bs < 16)
                s += vt[((size_t)img * 16 + ci) * 496 + a * 16 + bs] * w[(co * 16 + ci) * 7 + ks];
        }
    d4o[i] = s;
}

// inverse v-shear: out(y,x) = d4o[(f%31), (f/31)] with f = x*32+y; cat4 [24,32)
__global__ void mle_invv_kernel(const float* __restrict__ d4o, float* __restrict__ cat4)
{
    int i = blockIdx.x * blockDim.x + threadIdx.x;
    const int total = NIMG * 8 * 256;
    if (i >= total) return;
    int p = i % 256; int tmp = i / 256;
    int co = tmp % 8; int img = tmp / 8;
    int y = p / 16, x = p % 16;
    int f = x * 32 + y;  // < 496 always
    float v = d4o[((size_t)img * 8 + co) * 496 + (f % 31) * 16 + (f / 31)];
    cat4[((size_t)img * 32 + 24 + co) * 256 + p] = fmaxf(v * BN_SCALE, 0.0f);
}

// c3: 1x1 conv 32->64, relu(bn), into cat1 (f16 channel-last) channels [64,128)
__global__ void mle_c3_kernel(const float* __restrict__ cat4, const float* __restrict__ w,
                              const float* __restrict__ bia, _Float16* __restrict__ cat1)
{
    int i = blockIdx.x * blockDim.x + threadIdx.x;
    const int total = NIMG * 64 * 256;
    if (i >= total) return;
    int p = i % 256; int tmp = i / 256;
    int co = tmp % 64; int img = tmp / 64;
    float s = bia[co];
    for (int ci = 0; ci < 32; ++ci)
        s += cat4[((size_t)img * 32 + ci) * 256 + p] * w[co * 32 + ci];
    cat1[((size_t)img * 256 + p) * 128 + 64 + co] = (_Float16)fmaxf(s * BN_SCALE, 0.0f);
}

// ---------------------------------------------------------------------------
// cat2 (f16 channel-last [b][row][col][128]):
//   ch<64 from x; ch>=64 deterministic scatter-average of x2 windows.
// ---------------------------------------------------------------------------
__global__ void cat2_build_kernel(const float* __restrict__ x,
                                  const _Float16* __restrict__ x2h,
                                  const int* __restrict__ sxA, const int* __restrict__ syA,
                                  _Float16* __restrict__ cat2)
{
    int i = blockIdx.x * blockDim.x + threadIdx.x;
    const int total = BB * HH * WW * 128;
    if (i >= total) return;
    int ch = i % 128; int tmp = i / 128;
    int col = tmp % WW; tmp /= WW;
    int row = tmp % HH; int b = tmp / HH;
    float val;
    if (ch < DIM) {
        val = x[(((size_t)b * DIM + ch) * HH + row) * WW + col];
    } else {
        int c2 = ch - DIM;
        float acc = 0.f, cnt = 0.f;
        for (int t = 0; t < KEEP2; ++t) {
            int sy = syA[b * KEEP2 + t], sx = sxA[b * KEEP2 + t];
            int u = row - sy, v = col - sx;
            if (u >= 0 && u < WIN && v >= 0 && v < WIN) {
                acc += (float)x2h[(((size_t)(b * KEEP2 + t) * WIN + u) * WIN + v) * DIM + c2];
                cnt += 1.0f;
            }
        }
        val = acc / (cnt + 1e-10f);
    }
    cat2[i] = (_Float16)val;
}

// ---------------------------------------------------------------------------
// Host side
// ---------------------------------------------------------------------------
static inline int cdiv(int a, int b) { return (a + b - 1) / b; }

extern "C" void kernel_launch(void* const* d_in, const int* in_sizes, int n_in,
                              void* d_out, int out_size, void* d_ws, size_t ws_size,
                              hipStream_t stream)
{
    (void)in_sizes; (void)n_in; (void)out_size; (void)ws_size;
    // Inputs in setup_inputs() dict insertion order (nested dicts flattened in
    // insertion order): x, prob, y, epoch, fix_w, conv1_0{w1,b1,w2,b2},
    // mle{c1w,c1b,d1w,d1b,d2w,d2b,d3w,d3b,d4w,d4b,c3w,c3b},
    // conv1{w1,b1,w2,b2}, conv2{w1,b1,w2,b2}
    const float* X      = (const float*)d_in[0];
    const float* PROB   = (const float*)d_in[1];
    const float* YV     = (const float*)d_in[2];
    /* d_in[3] = epoch (unused) */
    const float* FIXW   = (const float*)d_in[4];
    const float* C10_W1 = (const float*)d_in[5];
    const float* C10_B1 = (const float*)d_in[6];
    const float* C10_W2 = (const float*)d_in[7];
    const float* C10_B2 = (const float*)d_in[8];
    const float* M_C1W  = (const float*)d_in[9];
    const float* M_C1B  = (const float*)d_in[10];
    const float* M_D1W  = (const float*)d_in[11];
    const float* M_D1B  = (const float*)d_in[12];
    const float* M_D2W  = (const float*)d_in[13];
    const float* M_D2B  = (const float*)d_in[14];
    const float* M_D3W  = (const float*)d_in[15];
    const float* M_D3B  = (const float*)d_in[16];
    const float* M_D4W  = (const float*)d_in[17];
    const float* M_D4B  = (const float*)d_in[18];
    const float* M_C3W  = (const float*)d_in[19];
    const float* M_C3B  = (const float*)d_in[20];
    const float* C1_W1  = (const float*)d_in[21];
    const float* C1_B1  = (const float*)d_in[22];
    const float* C1_W2  = (const float*)d_in[23];
    const float* C1_B2  = (const float*)d_in[24];
    const float* C2_W1  = (const float*)d_in[25];
    const float* C2_B1  = (const float*)d_in[26];
    const float* C2_W2  = (const float*)d_in[27];
    const float* C2_B2  = (const float*)d_in[28];
    float* OUT = (float*)d_out;

    // Workspace bump allocator (~75 MB required).
    char* base = (char*)d_ws;
    size_t off = 0;
    auto alloc = [&](size_t bytes) -> char* {
        off = (off + 255) & ~(size_t)255;
        char* p = base + off;
        off += bytes;
        return p;
    };
    float*     ent   = (float*)    alloc((size_t)BB * HPD * WPD * 4);
    float*     enty  = (float*)    alloc((size_t)BB * HPD * WPD * 4);
    float*     score = (float*)    alloc((size_t)BB * NBOX * 4);
    int*       keep1 = (int*)      alloc((size_t)BB * KEEP1 * 4);
    float*     wsc   = (float*)    alloc((size_t)BB * KEEP1 * 4);
    int*       sxA   = (int*)      alloc((size_t)BB * KEEP2 * 4);
    int*       syA   = (int*)      alloc((size_t)BB * KEEP2 * 4);
    float*     wf32  = (float*)    alloc((size_t)NIMG * DIM * 256 * 4);
    _Float16*  wf16  = (_Float16*) alloc((size_t)NIMG * 256 * DIM * 2);
    _Float16*  tmp16 = (_Float16*) alloc((size_t)NIMG * 256 * DIM * 2);
    _Float16*  cat1  = (_Float16*) alloc((size_t)NIMG * 256 * 128 * 2);
    float*     xc    = (float*)    alloc((size_t)NIMG * 16 * 256 * 4);
    float*     cat4  = (float*)    alloc((size_t)NIMG * 32 * 256 * 4);
    float*     ht    = (float*)    alloc((size_t)NIMG * 16 * 496 * 4);
    float*     d3o   = (float*)    alloc((size_t)NIMG * 8 * 496 * 4);
    float*     vt    = (float*)    alloc((size_t)NIMG * 16 * 496 * 4);
    float*     d4o   = (float*)    alloc((size_t)NIMG * 8 * 496 * 4);
    _Float16*  x2h   = (_Float16*) alloc((size_t)NIMG * 256 * DIM * 2);
    _Float16*  cat2  = (_Float16*) alloc((size_t)BB * HH * WW * 128 * 2);
    _Float16*  t2    = (_Float16*) alloc((size_t)BB * HH * WW * DIM * 2);
    _Float16*  c10w1p = (_Float16*)alloc((size_t)DIM * DIM * 9 * 2);
    _Float16*  c10w2p = (_Float16*)alloc((size_t)DIM * DIM * 9 * 2);
    _Float16*  c1w1p  = (_Float16*)alloc((size_t)DIM * 128 * 9 * 2);
    _Float16*  c1w2p  = (_Float16*)alloc((size_t)DIM * DIM * 9 * 2);
    _Float16*  c2w1p  = (_Float16*)alloc((size_t)DIM * 128 * 9 * 2);
    _Float16*  c2w2p  = (_Float16*)alloc((size_t)DIM * DIM * 9 * 2);

    const int TB = 256;

    // 0. pre-pack all 3x3 conv weights into B-fragment layout (f16)
    pack_weights_kernel<<<cdiv(DIM * DIM * 9, TB), TB, 0, stream>>>(C10_W1, c10w1p, DIM, DIM);
    pack_weights_kernel<<<cdiv(DIM * DIM * 9, TB), TB, 0, stream>>>(C10_W2, c10w2p, DIM, DIM);
    pack_weights_kernel<<<cdiv(DIM * 128 * 9, TB), TB, 0, stream>>>(C1_W1, c1w1p, DIM, 128);
    pack_weights_kernel<<<cdiv(DIM * DIM * 9, TB), TB, 0, stream>>>(C1_W2, c1w2p, DIM, DIM);
    pack_weights_kernel<<<cdiv(DIM * 128 * 9, TB), TB, 0, stream>>>(C2_W1, c2w1p, DIM, 128);
    pack_weights_kernel<<<cdiv(DIM * DIM * 9, TB), TB, 0, stream>>>(C2_W2, c2w2p, DIM, DIM);

    // 1. entropy maps
    entropy_kernel<<<cdiv(BB * HPD * WPD, TB), TB, 0, stream>>>(PROB, YV, ent, enty);
    // 2. window scores
    window_score_kernel<<<cdiv(BB * NBOX, TB), TB, 0, stream>>>(ent, FIXW, score);
    // 3. NMS pass 1 (keep 120)
    nms1_kernel<<<BB, 256, 0, stream>>>(score, keep1);
    // 4. crop re-score + NMS pass 2 (keep 30)
    crop_score_kernel<<<cdiv(BB * KEEP1, TB), TB, 0, stream>>>(enty, FIXW, keep1, wsc);
    nms2_kernel<<<BB, 128, 0, stream>>>(wsc, keep1, sxA, syA);
    // 5. gather windows (fp32 NCHW for MLE, f16 NHWC for WMMA)
    gather_windows32_kernel<<<cdiv(NIMG * DIM * 256, TB), TB, 0, stream>>>(X, sxA, syA, wf32);
    gather_windows16_kernel<<<cdiv(NIMG * 256 * DIM, TB), TB, 0, stream>>>(X, sxA, syA, wf16);

    // 6. conv1_0 double conv (WMMA): wf16 -> tmp16 -> cat1[:, :, 0:64]
    {
        dim3 g(16, 1, NIMG);   // H=16, W=16 -> 16 pixel tiles; all 64 couts per wave
        conv3x3_wmma_kernel<64, 16, 16, false><<<g, 32, 0, stream>>>(
            wf16, c10w1p, C10_B1, tmp16, 64, 0);
        conv3x3_wmma_kernel<64, 16, 16, false><<<g, 32, 0, stream>>>(
            tmp16, c10w2p, C10_B2, cat1, 128, 0);
    }
    // 7. MLE block -> cat1[:, :, 64:128]
    mle_c1_kernel<<<cdiv(NIMG * 16 * 256, TB), TB, 0, stream>>>(wf32, M_C1W, M_C1B, xc);
    mle_d1_kernel<<<cdiv(NIMG * 8 * 256, TB), TB, 0, stream>>>(xc, M_D1W, M_D1B, cat4);
    mle_d2_kernel<<<cdiv(NIMG * 8 * 256, TB), TB, 0, stream>>>(xc, M_D2W, M_D2B, cat4);
    htr_kernel<<<cdiv(NIMG * 16 * 496, TB), TB, 0, stream>>>(xc, ht);
    mle_d3_kernel<<<cdiv(NIMG * 8 * 496, TB), TB, 0, stream>>>(ht, M_D3W, M_D3B, d3o);
    mle_invh_kernel<<<cdiv(NIMG * 8 * 256, TB), TB, 0, stream>>>(d3o, cat4);
    vtr_kernel<<<cdiv(NIMG * 16 * 496, TB), TB, 0, stream>>>(xc, vt);
    mle_d4_kernel<<<cdiv(NIMG * 8 * 496, TB), TB, 0, stream>>>(vt, M_D4W, M_D4B, d4o);
    mle_invv_kernel<<<cdiv(NIMG * 8 * 256, TB), TB, 0, stream>>>(d4o, cat4);
    mle_c3_kernel<<<cdiv(NIMG * 64 * 256, TB), TB, 0, stream>>>(cat4, M_C3W, M_C3B, cat1);

    // 8. conv1 double conv (WMMA): cat1 (Cin=128) -> tmp16 -> x2h
    {
        dim3 g(16, 1, NIMG);
        conv3x3_wmma_kernel<128, 16, 16, false><<<g, 32, 0, stream>>>(
            cat1, c1w1p, C1_B1, tmp16, 64, 0);
        conv3x3_wmma_kernel<64, 16, 16, false><<<g, 32, 0, stream>>>(
            tmp16, c1w2p, C1_B2, x2h, 64, 0);
    }

    // 9. cat2 = concat(x, scatter-average(x2)) in f16 channel-last
    cat2_build_kernel<<<cdiv(BB * HH * WW * 128, TB), TB, 0, stream>>>(
        X, x2h, sxA, syA, cat2);

    // 10. conv2 double conv (WMMA, dominant cost): cat2 -> t2 -> OUT (fp32 NCHW)
    {
        dim3 g(HH * (WW / 16), 1, BB);   // 4096 pixel tiles x 2 images
        conv3x3_wmma_kernel<128, 256, 256, false><<<g, 32, 0, stream>>>(
            cat2, c2w1p, C2_B1, t2, 64, 0);
        conv3x3_wmma_kernel<64, 256, 256, true><<<g, 32, 0, stream>>>(
            t2, c2w2p, C2_B2, OUT, 64, 0);
    }
}
